// BiLSTMSeq2Seq_81612968559241
// MI455X (gfx1250) — matrix operand order repaired and checked by reading, not compiled, not run on hardware
//
#include <hip/hip_runtime.h>
#include <hip/hip_bf16.h>
#include <math.h>

// ---------------- model dims ----------------
#define V_SZ   50257
#define E_SZ   512
#define H_SZ   1024
#define S_SZ   2048
#define G_SZ   4096          // 4*H
#define T_DEC  128           // max_seq_len
#define XK     2560          // E + 2H (decoder input width)
#define BOS_TOK 1

typedef __bf16 bf16;
typedef __attribute__((ext_vector_type(16))) __bf16 v16bf;
typedef __attribute__((ext_vector_type(8)))  float  v8f;

// ---------------- WMMA helpers (V_WMMA_F32_16X16X32_BF16) ----------------
// A-fragment: lane l<16 -> row m0+(l&15), contiguous elems K = base..base+15,
// base = k0 + 16*(l>>4). For matvec we pass the same row pointer to every lane
// (all 16 A rows identical -> all C rows identical; read row 0).
__device__ __forceinline__ v16bf load_a_row(const bf16* __restrict__ row, int k0, int hf) {
    int base = k0 + hf * 16;
    v16bf a;
#pragma unroll
    for (int i = 0; i < 16; ++i) a[i] = row[base + i];
    return a;
}
// B-fragment: lane l<16 -> column n0+(l&15) = weight row (x@W.T), contiguous
// K = k0 + 16*(l>>4) .. +15.
__device__ __forceinline__ v16bf load_b_col(const bf16* __restrict__ wrow, int k0, int hf) {
    int kb = k0 + hf * 16;
    v16bf b;
#pragma unroll
    for (int i = 0; i < 16; ++i) b[i] = wrow[kb + i];
    return b;
}
__device__ __forceinline__ v8f wmma_bf16(v16bf a, v16bf b, v8f c) {
    return __builtin_amdgcn_wmma_f32_16x16x32_bf16(false, a, false, b, (short)0, c, false, false);
}
__device__ __forceinline__ float sigmoidf_(float x) { return 1.0f / (1.0f + __expf(-x)); }

// ---------------- kernels ----------------
__global__ void k_f2bf(const float* __restrict__ src, bf16* __restrict__ dst, int n) {
    for (int i = blockIdx.x * blockDim.x + threadIdx.x; i < n; i += gridDim.x * blockDim.x)
        dst[i] = (bf16)src[i];
}

__global__ void k_gather(const int* __restrict__ inputs, const float* __restrict__ emb,
                         bf16* __restrict__ X, float* __restrict__ mask) {
    const int total = S_SZ * E_SZ;
    for (int i = blockIdx.x * blockDim.x + threadIdx.x; i < total; i += gridDim.x * blockDim.x) {
        int s = i >> 9, e = i & (E_SZ - 1);
        int tok = inputs[s];
        X[i] = (bf16)emb[(size_t)tok * E_SZ + e];
        if (e == 0) mask[s] = (tok != 0) ? 1.0f : 0.0f;
    }
}

// P[s,:] = X[s,:] @ Wih.T + b   (M=2048, N=4096, K=512) ; one 16x16 tile per wave
__global__ void k_preproj(const bf16* __restrict__ X, const bf16* __restrict__ Wih,
                          const float* __restrict__ bias, float* __restrict__ P) {
    int wave = blockIdx.x * (blockDim.x >> 5) + (threadIdx.x >> 5);
    int lane = threadIdx.x & 31, hf = lane >> 4, lq = lane & 15;
    const int NT = G_SZ / 16;                  // 256 column tiles
    int m0 = (wave / NT) * 16, n0 = (wave % NT) * 16;
    const bf16* arow = X   + (size_t)(m0 + lq) * E_SZ;
    const bf16* brow = Wih + (size_t)(n0 + lq) * E_SZ;
    v8f acc = {};
    for (int k0 = 0; k0 < E_SZ; k0 += 32)
        acc = wmma_bf16(load_a_row(arow, k0, hf), load_b_col(brow, k0, hf), acc);
    float bs = bias[n0 + lq];
    float* pout = P + (size_t)m0 * G_SZ + n0 + lq;
#pragma unroll
    for (int i = 0; i < 8; ++i) pout[(size_t)(i + hf * 8) * G_SZ] = acc[i] + bs;
}

// Sequential LSTM over 2048 steps; blockIdx.x = direction (0=fwd, 1=bwd).
// 32 waves x 8 N-tiles = 4096 gate outputs per step via broadcast-A WMMA matvec.
__global__ void k_enc_rnn(const float* __restrict__ Pf, const float* __restrict__ Pb,
                          const bf16* __restrict__ Whf, const bf16* __restrict__ Whb,
                          const float* __restrict__ mask, bf16* __restrict__ enc_state) {
    __shared__ __align__(32) float h_s[H_SZ];
    __shared__ __align__(32) float c_s[H_SZ];
    __shared__ __align__(32) bf16  hb[H_SZ];
    __shared__ __align__(32) float g_s[G_SZ];
    const int dir = blockIdx.x;
    const float* P  = dir ? Pb : Pf;
    const bf16* Whh = dir ? Whb : Whf;
    int tid = threadIdx.x, wave = tid >> 5, lane = tid & 31, hf = lane >> 4, lq = lane & 15;
    for (int j = tid; j < H_SZ; j += blockDim.x) { h_s[j] = 0.f; c_s[j] = 0.f; hb[j] = (bf16)0.0f; }
    __syncthreads();
    for (int t = 0; t < S_SZ; ++t) {
        int s = dir ? (S_SZ - 1 - t) : t;
        const float* prow = P + (size_t)s * G_SZ;
        for (int tt = 0; tt < 8; ++tt) {
            int n0 = (wave * 8 + tt) * 16;
            const bf16* brow = Whh + (size_t)(n0 + lq) * H_SZ;
            v8f acc = {};
            for (int k0 = 0; k0 < H_SZ; k0 += 32)
                acc = wmma_bf16(load_a_row(hb, k0, hf), load_b_col(brow, k0, hf), acc);
            if (hf == 0) g_s[n0 + lq] = prow[n0 + lq] + acc[0];
        }
        __syncthreads();
        float m = mask[s];
        for (int j = tid; j < H_SZ; j += blockDim.x) {
            float ig = g_s[j], fg = g_s[H_SZ + j], gg = g_s[2 * H_SZ + j], og = g_s[3 * H_SZ + j];
            float cn = sigmoidf_(fg) * c_s[j] + sigmoidf_(ig) * tanhf(gg);
            float hn = sigmoidf_(og) * tanhf(cn);
            float hm = m * hn + (1.f - m) * h_s[j];
            float cm = m * cn + (1.f - m) * c_s[j];
            h_s[j] = hm; c_s[j] = cm; hb[j] = (bf16)hm;
        }
        __syncthreads();
    }
    for (int j = tid; j < H_SZ; j += blockDim.x)
        enc_state[dir * H_SZ + j] = (bf16)h_s[j];
}

__global__ void k_dec_init(float* __restrict__ h, float* __restrict__ c,
                           bf16* __restrict__ hb, int* __restrict__ tok) {
    int j = threadIdx.x;
    if (j < H_SZ) { h[j] = 0.f; c[j] = 0.f; hb[j] = (bf16)0.0f; }
    if (j == 0) *tok = BOS_TOK;
}

// One decoder LSTM step: gates over K = 2560 (x part) + 1024 (h part), then h/c update.
__global__ void k_dec_gates(const int* __restrict__ tok_p, const float* __restrict__ dec_emb,
                            const bf16* __restrict__ enc_state,
                            const bf16* __restrict__ Wih, const bf16* __restrict__ Whh,
                            const float* __restrict__ bias,
                            float* __restrict__ h_g, float* __restrict__ c_g,
                            bf16* __restrict__ hb_g) {
    __shared__ __align__(32) bf16  x_s[XK];
    __shared__ __align__(32) bf16  hb[H_SZ];
    __shared__ __align__(32) float g_s[G_SZ];
    int tid = threadIdx.x, wave = tid >> 5, lane = tid & 31, hf = lane >> 4, lq = lane & 15;
    int tok = *tok_p;
    for (int j = tid; j < E_SZ; j += blockDim.x)     x_s[j] = (bf16)dec_emb[(size_t)tok * E_SZ + j];
    for (int j = tid; j < 2 * H_SZ; j += blockDim.x) x_s[E_SZ + j] = enc_state[j];
    for (int j = tid; j < H_SZ; j += blockDim.x)     hb[j] = hb_g[j];
    __syncthreads();
    for (int tt = 0; tt < 8; ++tt) {
        int n0 = (wave * 8 + tt) * 16;
        const bf16* bwih = Wih + (size_t)(n0 + lq) * XK;
        const bf16* bwhh = Whh + (size_t)(n0 + lq) * H_SZ;
        v8f acc = {};
        for (int k0 = 0; k0 < XK; k0 += 32)
            acc = wmma_bf16(load_a_row(x_s, k0, hf), load_b_col(bwih, k0, hf), acc);
        for (int k0 = 0; k0 < H_SZ; k0 += 32)
            acc = wmma_bf16(load_a_row(hb, k0, hf), load_b_col(bwhh, k0, hf), acc);
        if (hf == 0) g_s[n0 + lq] = acc[0] + bias[n0 + lq];
    }
    __syncthreads();
    for (int j = tid; j < H_SZ; j += blockDim.x) {
        float ig = g_s[j], fg = g_s[H_SZ + j], gg = g_s[2 * H_SZ + j], og = g_s[3 * H_SZ + j];
        float cn = sigmoidf_(fg) * c_g[j] + sigmoidf_(ig) * tanhf(gg);
        float hn = sigmoidf_(og) * tanhf(cn);
        c_g[j] = cn; h_g[j] = hn; hb_g[j] = (bf16)hn;
    }
}

// logits[n] = h . out_W[n,:] + out_b[n]; one 16-vocab tile per wave (bf16 out_W ~103MB -> L2 resident)
__global__ void k_dec_logits(const bf16* __restrict__ hb_g, const bf16* __restrict__ Wout,
                             const float* __restrict__ bout, float* __restrict__ out_row) {
    __shared__ __align__(32) bf16 hb[H_SZ];
    int tid = threadIdx.x, wave = tid >> 5, lane = tid & 31, hf = lane >> 4, lq = lane & 15;
    for (int j = tid; j < H_SZ; j += blockDim.x) hb[j] = hb_g[j];
    __syncthreads();
    int n0 = (blockIdx.x * (blockDim.x >> 5) + wave) * 16;
    if (n0 >= V_SZ) return;                        // wave-uniform exit, EXEC stays full for WMMA
    int n  = n0 + lq;
    int nr = (n < V_SZ) ? n : (V_SZ - 1);          // clamp tail rows (writes masked below)
    const bf16* brow = Wout + (size_t)nr * H_SZ;
    v8f acc = {};
    for (int k0 = 0; k0 < H_SZ; k0 += 32)
        acc = wmma_bf16(load_a_row(hb, k0, hf), load_b_col(brow, k0, hf), acc);
    if (hf == 0 && n < V_SZ) out_row[n] = acc[0] + bout[n];
}

__global__ void k_argmax(const float* __restrict__ logits, int* __restrict__ tok_p,
                         float* __restrict__ seq_out) {
    __shared__ float smax[1024];
    __shared__ int   sidx[1024];
    int tid = threadIdx.x;
    float best = -INFINITY; int bi = 0;
    for (int v = tid; v < V_SZ; v += 1024) {
        float x = logits[v];
        if (x > best) { best = x; bi = v; }        // strided ascending -> first max per thread
    }
    smax[tid] = best; sidx[tid] = bi;
    __syncthreads();
    for (int s = 512; s > 0; s >>= 1) {
        if (tid < s) {
            float a = smax[tid], b = smax[tid + s];
            int ia = sidx[tid], ib = sidx[tid + s];
            if (b > a || (b == a && ib < ia)) { smax[tid] = b; sidx[tid] = ib; }
        }
        __syncthreads();
    }
    if (tid == 0) { *tok_p = sidx[0]; *seq_out = (float)sidx[0]; }
}

// ---------------- host ----------------
extern "C" void kernel_launch(void* const* d_in, const int* in_sizes, int n_in,
                              void* d_out, int out_size, void* d_ws, size_t ws_size,
                              hipStream_t stream) {
    (void)in_sizes; (void)n_in; (void)out_size; (void)ws_size;
    const int*   inputs  = (const int*)d_in[0];
    const float* enc_emb = (const float*)d_in[2];
    const float* eWihF = (const float*)d_in[3];
    const float* eWhhF = (const float*)d_in[4];
    const float* ebF   = (const float*)d_in[5];
    const float* eWihB = (const float*)d_in[6];
    const float* eWhhB = (const float*)d_in[7];
    const float* ebB   = (const float*)d_in[8];
    const float* demb  = (const float*)d_in[9];
    const float* dWih  = (const float*)d_in[10];
    const float* dWhh  = (const float*)d_in[11];
    const float* dbv   = (const float*)d_in[12];
    const float* outW  = (const float*)d_in[13];
    const float* outb  = (const float*)d_in[14];
    float* out = (float*)d_out;

    char* ws = (char*)d_ws;
    size_t off = 0;
    auto alloc = [&](size_t bytes) -> char* {
        char* p = ws + off; off = (off + bytes + 255) & ~(size_t)255; return p;
    };
    bf16* w_eih_f = (bf16*)alloc((size_t)G_SZ * E_SZ * 2);
    bf16* w_ehh_f = (bf16*)alloc((size_t)G_SZ * H_SZ * 2);
    bf16* w_eih_b = (bf16*)alloc((size_t)G_SZ * E_SZ * 2);
    bf16* w_ehh_b = (bf16*)alloc((size_t)G_SZ * H_SZ * 2);
    bf16* w_dih   = (bf16*)alloc((size_t)G_SZ * XK   * 2);
    bf16* w_dhh   = (bf16*)alloc((size_t)G_SZ * H_SZ * 2);
    bf16* w_out   = (bf16*)alloc((size_t)V_SZ * H_SZ * 2);
    bf16* Xb      = (bf16*)alloc((size_t)S_SZ * E_SZ * 2);
    bf16* encst   = (bf16*)alloc((size_t)2 * H_SZ * 2);
    bf16* hb_dec  = (bf16*)alloc((size_t)H_SZ * 2);
    float* maskf  = (float*)alloc((size_t)S_SZ * 4);
    float* Pf     = (float*)alloc((size_t)S_SZ * G_SZ * 4);
    float* Pb     = (float*)alloc((size_t)S_SZ * G_SZ * 4);
    float* h_dec  = (float*)alloc((size_t)H_SZ * 4);
    float* c_dec  = (float*)alloc((size_t)H_SZ * 4);
    int*   tok    = (int*)alloc(4);

    auto cvt = [&](const float* s, bf16* d, int n) {
        int blocks = (n + 255) / 256; if (blocks > 4096) blocks = 4096;
        k_f2bf<<<blocks, 256, 0, stream>>>(s, d, n);
    };
    cvt(eWihF, w_eih_f, G_SZ * E_SZ);
    cvt(eWhhF, w_ehh_f, G_SZ * H_SZ);
    cvt(eWihB, w_eih_b, G_SZ * E_SZ);
    cvt(eWhhB, w_ehh_b, G_SZ * H_SZ);
    cvt(dWih,  w_dih,   G_SZ * XK);
    cvt(dWhh,  w_dhh,   G_SZ * H_SZ);
    cvt(outW,  w_out,   V_SZ * H_SZ);

    k_gather<<<2048, 256, 0, stream>>>(inputs, enc_emb, Xb, maskf);

    // 128 row-tiles * 256 col-tiles = 32768 waves, 8 waves/block
    k_preproj<<<4096, 256, 0, stream>>>(Xb, w_eih_f, ebF, Pf);
    k_preproj<<<4096, 256, 0, stream>>>(Xb, w_eih_b, ebB, Pb);

    k_enc_rnn<<<2, 1024, 0, stream>>>(Pf, Pb, w_ehh_f, w_ehh_b, maskf, encst);

    k_dec_init<<<1, 1024, 0, stream>>>(h_dec, c_dec, hb_dec, tok);

    const int LOGIT_BLOCKS = (((V_SZ + 15) / 16) + 7) / 8;   // 393
    for (int t = 0; t < T_DEC; ++t) {
        float* lrow = out + T_DEC + (size_t)t * V_SZ;
        k_dec_gates<<<1, 1024, 0, stream>>>(tok, demb, encst, w_dih, w_dhh, dbv,
                                            h_dec, c_dec, hb_dec);
        k_dec_logits<<<LOGIT_BLOCKS, 256, 0, stream>>>(hb_dec, w_out, outb, lrow);
        k_argmax<<<1, 1024, 0, stream>>>(lrow, tok, out + t);
    }
}
